// TuckER_22136261444361
// MI455X (gfx1250) — matrix operand order repaired and checked by reading, not compiled, not run on hardware
//
#include <hip/hip_runtime.h>
#include <math.h>

typedef __attribute__((ext_vector_type(16))) __bf16 bf16x16;
typedef __attribute__((ext_vector_type(8)))  float  v8f;
typedef int i32x4 __attribute__((vector_size(16)));   // matches builtin param type

union ABArg {
    bf16x16 v;
    uint4   u[2];
};

#if defined(__has_builtin)
# if __has_builtin(__builtin_amdgcn_global_load_async_to_lds_b128)
#  define HAVE_ASYNC_LDS 1
# else
#  define HAVE_ASYNC_LDS 0
# endif
# if __has_builtin(__builtin_amdgcn_s_wait_asynccnt)
#  define HAVE_WAIT_ASYNC 1
# else
#  define HAVE_WAIT_ASYNC 0
# endif
#else
# define HAVE_ASYNC_LDS 0
# define HAVE_WAIT_ASYNC 0
#endif

#define GLOBAL_AS __attribute__((address_space(1)))
#define LDS_AS    __attribute__((address_space(3)))

// copy 16 bytes global -> LDS (async if the gfx1250 builtin is available)
__device__ __forceinline__ void async_copy16(const unsigned short* g, unsigned short* l) {
#if HAVE_ASYNC_LDS
    __builtin_amdgcn_global_load_async_to_lds_b128(
        (GLOBAL_AS i32x4*)(uintptr_t)g,
        (LDS_AS i32x4*)(unsigned int)(uintptr_t)l,
        0, 0);
#else
    *(uint4*)l = *(const uint4*)g;
#endif
}

__device__ __forceinline__ void wait_async_all() {
#if HAVE_ASYNC_LDS
# if HAVE_WAIT_ASYNC
    __builtin_amdgcn_s_wait_asynccnt(0);
# else
    asm volatile("s_wait_asynccnt 0x0" ::: "memory");
# endif
#endif
}

__device__ __forceinline__ unsigned short f32_to_bf16(float f) {
    unsigned int u = __float_as_uint(f);
    unsigned int r = u + 0x7FFFu + ((u >> 16) & 1u);   // round-to-nearest-even
    return (unsigned short)(r >> 16);
}

// ---------------------------------------------------------------------------
// init: zero the 4096 accumulators and the 3 norm accumulators
// ---------------------------------------------------------------------------
__global__ void tucker_init(float* __restrict__ out, float* __restrict__ sums) {
    int t = blockIdx.x * 256 + threadIdx.x;
    if (t < 4096) out[t] = 0.0f;
    if (blockIdx.x == 0 && threadIdx.x < 3) sums[threadIdx.x] = 0.0f;
}

// ---------------------------------------------------------------------------
// convert W (256^3 fp32) -> bf16, vectorized 4-at-a-time
// ---------------------------------------------------------------------------
__global__ void tucker_convW(const float* __restrict__ W,
                             unsigned short* __restrict__ Wbf) {
    int t = blockIdx.x * blockDim.x + threadIdx.x;   // one float4 per thread
    float4 f = ((const float4*)W)[t];
    ushort4 o;
    o.x = f32_to_bf16(f.x);
    o.y = f32_to_bf16(f.y);
    o.z = f32_to_bf16(f.z);
    o.w = f32_to_bf16(f.w);
    ((ushort4*)Wbf)[t] = o;
}

// ---------------------------------------------------------------------------
// gather q rows and convert to bf16: qbf[b][d] = bf16(Q_table[qs[b]][d])
// ---------------------------------------------------------------------------
__global__ void tucker_gatherQ(const float* __restrict__ Q,
                               const int* __restrict__ qs,
                               unsigned short* __restrict__ qbf) {
    int b = blockIdx.x;
    int d = threadIdx.x;
    qbf[(size_t)b * 256 + d] = f32_to_bf16(Q[(size_t)qs[b] * 256 + d]);
}

// ---------------------------------------------------------------------------
// Frobenius-norm^2 accumulation for gathered p / q / r matrices
// ---------------------------------------------------------------------------
__global__ void tucker_norms(const float* __restrict__ P,
                             const float* __restrict__ Q,
                             const float* __restrict__ R,
                             const int* __restrict__ ps,
                             const int* __restrict__ qs,
                             const int* __restrict__ rs,
                             float* __restrict__ sums) {
    int which = blockIdx.y;           // 0=p, 1=q, 2=r
    int b = blockIdx.x;
    int d = threadIdx.x;
    const float* T  = (which == 0) ? P  : (which == 1) ? Q  : R;
    const int*   ix = (which == 0) ? ps : (which == 1) ? qs : rs;
    float v = T[(size_t)ix[b] * 256 + d];
    v *= v;
    #pragma unroll
    for (int o = 16; o > 0; o >>= 1) v += __shfl_xor(v, o, 32);
    if ((threadIdx.x & 31) == 0) atomicAdd(&sums[which], v);
}

__global__ void tucker_finalize(const float* __restrict__ sums,
                                float* __restrict__ out) {
    out[4096] = 0.01f * (sqrtf(sums[0]) + sqrtf(sums[1]) + sqrtf(sums[2]));
}

// ---------------------------------------------------------------------------
// Main contraction.
//   grid = (64 b-groups, 16 j-tiles), block = 128 (4 waves).
//   All 4 waves share j-tile jt; wave w owns b-tile bBase = bx*64 + w*16.
//   A slab W[i, jt*16..+15, 0..255] (8 KB, contiguous) is staged in LDS by
//   async global->LDS copies (2 i's per stage, double buffered, padded rows),
//   shared by the 4 waves -> 4x less L2 traffic for W.
//   A-fragment DS loads are software-pipelined one kk-step ahead so WMMAs
//   never wait on just-issued ds_load_b128s.
// ---------------------------------------------------------------------------
#define ROWSH 264   // padded row length in ushorts (528 B): conflict-free b128 reads

__global__ __launch_bounds__(128) void tucker_main(
    const unsigned short* __restrict__ Wbf,   // [256][256][256] bf16
    const unsigned short* __restrict__ qbf,   // [4096][256] bf16
    const int* __restrict__ ps,
    const int* __restrict__ rs,
    const float* __restrict__ P_table,
    const float* __restrict__ R_table,
    float* __restrict__ out) {

    // [buffer][i-within-stage][row][padded cols]
    __shared__ __align__(16) unsigned short Abuf[2][2][16][ROWSH];

    const int tid   = threadIdx.x;
    const int wave  = tid >> 5;
    const int lane  = tid & 31;
    const int jt    = blockIdx.y;                   // j-tile 0..15 (shared)
    const int bBase = blockIdx.x * 64 + wave * 16;  // per-wave b-tile

    const int n     = lane & 15;                    // b within tile (N index)
    const int khalf = lane >> 4;                    // lane half -> K sub-range

    // r weights matching the C/D layout: acc element v at this lane is
    // (j = jt*16 + v + 8*khalf, b = bBase + n)
    float rw[8];
    {
        const float* rrow =
            R_table + (size_t)rs[bBase + n] * 256 + jt * 16 + khalf * 8;
        #pragma unroll
        for (int v = 0; v < 8; ++v) rw[v] = rrow[v];
    }
    // per-lane p row base (cached in WGP$ across the whole i loop)
    const float* prow = P_table + (size_t)ps[bBase + n] * 256;

    // B fragments (q for this b-tile) are i-invariant: keep in 64 VGPRs
    const unsigned short* bptr = qbf + (size_t)(bBase + n) * 256 + khalf * 8;
    ABArg bfrag[8];
    #pragma unroll
    for (int kk = 0; kk < 8; ++kk) {
        bfrag[kk].u[0] = *(const uint4*)(bptr + kk * 32);
        bfrag[kk].u[1] = *(const uint4*)(bptr + kk * 32 + 16);
    }

    const unsigned short* slab0 = Wbf + ((size_t)jt << 12);  // + i*65536

    // ---- prefetch stage 0 (i = 0,1) into buffer 0 --------------------------
    #pragma unroll
    for (int c = 0; c < 8; ++c) {
        int u   = tid + c * 128;        // 0..1023 16-byte units (2 slabs)
        int ii  = u >> 9;               // which i of the stage
        int rem = u & 511;              // unit within 8 KB slab
        async_copy16(slab0 + ((size_t)ii << 16) + rem * 8,
                     &Abuf[0][ii][rem >> 5][(rem & 31) * 8]);
    }
    wait_async_all();
    __syncthreads();

    float acc = 0.0f;
    for (int i = 0; i < 256; i += 2) {
        const int cur = (i >> 1) & 1;

        // prefetch next stage (i+2, i+3) into the other buffer
        if (i + 2 < 256) {
            const unsigned short* src = slab0 + ((size_t)(i + 2) << 16);
            #pragma unroll
            for (int c = 0; c < 8; ++c) {
                int u   = tid + c * 128;
                int ii  = u >> 9;
                int rem = u & 511;
                async_copy16(src + ((size_t)ii << 16) + rem * 8,
                             &Abuf[cur ^ 1][ii][rem >> 5][(rem & 31) * 8]);
            }
        }

        // two independent WMMA accumulation chains (i and i+1);
        // A fragments loaded one kk-step ahead of consumption.
        const unsigned short* a0 = &Abuf[cur][0][n][khalf * 8];
        const unsigned short* a1 = &Abuf[cur][1][n][khalf * 8];
        v8f z0 = {}, z1 = {};
        ABArg f0, f1, g0, g1;
        f0.u[0] = *(const uint4*)(a0);
        f0.u[1] = *(const uint4*)(a0 + 16);
        f1.u[0] = *(const uint4*)(a1);
        f1.u[1] = *(const uint4*)(a1 + 16);
        #pragma unroll
        for (int kk = 0; kk < 8; ++kk) {        // K = 256 in steps of 32
            if (kk < 7) {
                g0.u[0] = *(const uint4*)(a0 + (kk + 1) * 32);
                g0.u[1] = *(const uint4*)(a0 + (kk + 1) * 32 + 16);
                g1.u[0] = *(const uint4*)(a1 + (kk + 1) * 32);
                g1.u[1] = *(const uint4*)(a1 + (kk + 1) * 32 + 16);
            }
            z0 = __builtin_amdgcn_wmma_f32_16x16x32_bf16(
                     false, f0.v, false, bfrag[kk].v, (short)0, z0, false, false);
            z1 = __builtin_amdgcn_wmma_f32_16x16x32_bf16(
                     false, f1.v, false, bfrag[kk].v, (short)0, z1, false, false);
            f0 = g0;
            f1 = g1;
        }

        float t0 = 0.0f, t1 = 0.0f;
        #pragma unroll
        for (int v = 0; v < 8; ++v) { t0 += rw[v] * z0[v]; t1 += rw[v] * z1[v]; }
        acc += prow[i] * t0 + prow[i + 1] * t1;   // fold in p[b,i]

        wait_async_all();      // slab for next stage has landed
        __syncthreads();       // and everyone is done reading this one
    }

    // lanes l and l+16 hold complementary j halves of the same b
    acc += __shfl_xor(acc, 16, 32);
    if (lane < 16) atomicAdd(&out[bBase + lane], acc);
}

// ---------------------------------------------------------------------------
extern "C" void kernel_launch(void* const* d_in, const int* in_sizes, int n_in,
                              void* d_out, int out_size, void* d_ws, size_t ws_size,
                              hipStream_t stream) {
    const int*   ps = (const int*)d_in[0];
    const int*   qs = (const int*)d_in[1];
    const int*   rs = (const int*)d_in[2];
    const float* P  = (const float*)d_in[3];
    const float* Q  = (const float*)d_in[4];
    const float* R  = (const float*)d_in[5];
    const float* W  = (const float*)d_in[6];
    float* out = (float*)d_out;

    // workspace layout: [0,32MB) W bf16 | [32MB,34MB) q bf16 | 3 floats sums
    unsigned short* Wbf  = (unsigned short*)d_ws;
    unsigned short* qbf  = Wbf + (size_t)16777216;
    float*          sums = (float*)((char*)d_ws + 33554432u + 2097152u);

    tucker_init<<<16, 256, 0, stream>>>(out, sums);
    tucker_convW<<<16384, 256, 0, stream>>>(W, Wbf);
    tucker_gatherQ<<<4096, 256, 0, stream>>>(Q, qs, qbf);
    tucker_norms<<<dim3(4096, 3), 256, 0, stream>>>(P, Q, R, ps, qs, rs, sums);
    tucker_finalize<<<1, 1, 0, stream>>>(sums, out);
    tucker_main<<<dim3(64, 16), 128, 0, stream>>>(Wbf, qbf, ps, rs, P, R, out);
}
